// Head_87660282511715
// MI455X (gfx1250) — matrix operand order repaired and checked by reading, not compiled, not run on hardware
//
#include <hip/hip_runtime.h>
#include <hip/hip_bf16.h>
#include <stdint.h>

// ------------------------------------------------------------------
// CDNA5 (gfx1250) vector types for WMMA
// ------------------------------------------------------------------
typedef __attribute__((ext_vector_type(16))) __bf16 v16bf;
typedef __attribute__((ext_vector_type(8)))  float  v8f;

union FragBF {        // 32B fragment = 16 bf16 = two 16B LDS loads
  v16bf  v;
  float4 q[2];
};

// ------------------------------------------------------------------
// Problem dims
// ------------------------------------------------------------------
#define CD     512        // feature dim (GEMM K)
#define MR     8192       // query rows  (8*32*32)
#define NR     16384      // database rows
#define KTOP   9
#define BATCH  8
#define HW     1024       // 32*32 patches per batch
#define IMG    256        // output image side

// GEMM tiling
#define MT     64         // M rows per workgroup
#define NT     128        // N cols per tile
#define KT     32         // K per WMMA step
#define NTILES (NR / NT)  // 128
#define KSTEPS (CD / KT)  // 16

// Static LDS layout (bytes)
#define SA_OFF     0                        // 2 * 64*32*2  = 8192
#define SB_OFF     8192                     // 2 * 128*32*2 = 16384
#define SD_OFF     (8192 + 16384)           // 64*128*4     = 32768
#define SNB_OFF    (8192 + 16384 + 32768)   // 128*4        = 512
#define SMEM_BYTES (SNB_OFF + NT * 4)       // 57856 < 64KB

// ------------------------------------------------------------------
// CDNA5 async global->LDS copy (ASYNCcnt-tracked), per-lane b128.
// ------------------------------------------------------------------
static __device__ __forceinline__ void async_load_b128(void* lds, const void* gaddr) {
  asm volatile("global_load_async_to_lds_b128 %0, %1, off"
               :
               : "v"((uint32_t)(uintptr_t)lds),
                 "v"((uint64_t)(uintptr_t)gaddr)
               : "memory");
}
static __device__ __forceinline__ void wait_async_zero() {
  asm volatile("s_wait_asynccnt 0" ::: "memory");
}

static __device__ __forceinline__ int imin(int a, int b) { return a < b ? a : b; }
static __device__ __forceinline__ int imax(int a, int b) { return a > b ? a : b; }
static __device__ __forceinline__ int refl101(int i, int n) {
  i = (i < 0) ? -i : i;
  i = (i >= n) ? (2 * n - 2 - i) : i;
  return i;
}

// ------------------------------------------------------------------
// K1: f32 -> bf16 conversion + row squared-norm. One block per row.
// ------------------------------------------------------------------
__global__ __launch_bounds__(256)
void cvt_norm_kernel(const float* __restrict__ src, __bf16* __restrict__ dst,
                     float* __restrict__ nrm) {
  const int row = blockIdx.x;
  const int tid = threadIdx.x;
  const size_t base = (size_t)row * CD;
  float a0 = src[base + tid];
  float a1 = src[base + tid + 256];
  dst[base + tid]       = (__bf16)a0;
  dst[base + tid + 256] = (__bf16)a1;
  __shared__ float red[256];
  red[tid] = a0 * a0 + a1 * a1;
  __syncthreads();
  for (int s = 128; s > 0; s >>= 1) {
    if (tid < s) red[tid] += red[tid + s];
    __syncthreads();
  }
  if (tid == 0) nrm[row] = red[0];
}

// ------------------------------------------------------------------
// Chunk stagers: async global->LDS, double buffered.
//   A chunk: 64 rows x 32 K bf16 = 4KB  (1 b128 per thread)
//   B chunk: 128 rows x 32 K bf16 = 8KB (2 b128 per thread)
// Both stored row-major, 64B row stride.
// ------------------------------------------------------------------
static __device__ __forceinline__ void load_A_chunk(const __bf16* A, char* dst,
                                                    int m0, int k0, int tid) {
  int row = tid >> 2;
  int seg = tid & 3;
  const char* g = (const char*)(A + (size_t)(m0 + row) * CD + k0) + seg * 16;
  async_load_b128(dst + row * 64 + seg * 16, g);
}
static __device__ __forceinline__ void load_B_chunk(const __bf16* B, char* dst,
                                                    int n0, int k0, int tid) {
#pragma unroll
  for (int i = 0; i < 2; ++i) {
    int idx = tid + i * 256;
    int row = idx >> 2;
    int seg = idx & 3;
    const char* g = (const char*)(B + (size_t)(n0 + row) * CD + k0) + seg * 16;
    async_load_b128(dst + row * 64 + seg * 16, g);
  }
}

// ------------------------------------------------------------------
// K2: fused  -2*A.Bt GEMM (bf16 WMMA, f32 acc) + distance + top-9.
// Grid: 128 WGs x 256 threads (8 wave32).  Wave grid 4(M) x 2(N):
// each wave owns a 16x64 accumulator strip = 4 x v8f.
// Top-9 scan: 128 threads, 2 per row (64 cols each), merged once at
// the end with a constant-index sorted-merge network.
// ------------------------------------------------------------------
__global__ __launch_bounds__(256)
void fused_dist_topk_kernel(const __bf16* __restrict__ A, const __bf16* __restrict__ B,
                            const float* __restrict__ nAg, const float* __restrict__ nBg,
                            float* __restrict__ topkOut) {
  __shared__ __align__(32) char smem[SMEM_BYTES];
  char*  sA  = smem + SA_OFF;
  char*  sB  = smem + SB_OFF;
  float* sD  = (float*)(smem + SD_OFF);
  float* snB = (float*)(smem + SNB_OFF);

  const int tid  = threadIdx.x;
  const int lane = tid & 31;
  const int wave = tid >> 5;
  const int wm   = wave & 3;   // M sub-block (16 rows)
  const int wn   = wave >> 2;  // N sub-block (64 cols)
  const int m0   = blockIdx.x * MT;

  const int l16  = lane & 15;
  const int lhi  = lane >> 4;  // 0 for lanes 0-15, 1 for 16-31

  // scan assignment: threads 0..127 -> (row = tid&63, half = tid>>6)
  const int srow  = tid & 63;
  const int shalf = (tid >> 6) & 1;
  const float myNa = (tid < 2 * MT) ? nAg[m0 + srow] : 0.0f;

  float best[KTOP];
#pragma unroll
  for (int j = 0; j < KTOP; ++j) best[j] = 3.0e38f;

  // hoisted LDS fragment base addresses (per lane)
  const char* aFragBase = sA + (wm * 16 + l16) * 64 + (lhi << 4);
  const char* bFragBase = sB + (wn * 64 + l16) * 64 + (lhi << 4);

  // prologue: stage first A/B chunks into buffer 0
  load_A_chunk(A, sA, m0, 0, tid);
  load_B_chunk(B, sB, 0, 0, tid);
  wait_async_zero();
  __syncthreads();

  const v8f vzero = {};

  for (int nt = 0; nt < NTILES; ++nt) {
    const int n0 = nt * NT;
    if (tid < NT) snB[tid] = nBg[n0 + tid];

    v8f acc[4];
#pragma unroll
    for (int ai = 0; ai < 4; ++ai) acc[ai] = vzero;

    // keep a single copy of the body so the accumulators stay in one
    // fixed register tuple (WMMA accumulates in place, no v_mov churn)
#pragma clang loop unroll(disable)
    for (int ki = 0; ki < KSTEPS; ++ki) {
      const int cb = ki & 1;
      // issue next chunks into the other buffer (overlap with compute)
      if (!(nt == NTILES - 1 && ki == KSTEPS - 1)) {
        const int nki = (ki + 1) & (KSTEPS - 1);
        const int nn0 = (ki == KSTEPS - 1) ? (n0 + NT) : n0;
        load_A_chunk(A, sA + ((ki + 1) & 1) * 4096, m0, nki * KT, tid);
        load_B_chunk(B, sB + ((ki + 1) & 1) * 8192, nn0, nki * KT, tid);
      }

      // A fragment (16x32 bf16): lanes 0-15 hold K 0-7 & 16-23,
      // lanes 16-31 hold K 8-15 & 24-31 (ISA 05_wmma §7.12.2)
      FragBF fa;
      {
        const char* ap = aFragBase + cb * 4096;
        fa.q[0] = *(const float4*)(ap);
        fa.q[1] = *(const float4*)(ap + 32);
      }
#pragma unroll
      for (int ai = 0; ai < 4; ++ai) {
        // B fragment (32x16): lane = column, K half selected by lane>>4
        FragBF fb;
        const char* bp = bFragBase + cb * 8192 + ai * 1024;
        fb.q[0] = *(const float4*)(bp);
        fb.q[1] = *(const float4*)(bp + 32);
        acc[ai] = __builtin_amdgcn_wmma_f32_16x16x32_bf16(
            /*neg_a=*/false, fa.v, /*neg_b=*/false, fb.v,
            /*c_mod=*/(short)0, acc[ai], /*reuse_a=*/false, /*reuse_b=*/false);
      }

      wait_async_zero();   // next buffers resident (own wave's asyncs)
      __syncthreads();     // ... for everyone; prev buffer reads retired
    }

    // spill dot-products to LDS tile (C/D layout: VGPR j, lane l ->
    // M = j + 8*(l>>4), N = l&15)
#pragma unroll
    for (int ai = 0; ai < 4; ++ai) {
      const int col   = wn * 64 + ai * 16 + l16;
      const int rbase = wm * 16 + (lhi << 3);
#pragma unroll
      for (int j = 0; j < 8; ++j) sD[(rbase + j) * NT + col] = acc[ai][j];
    }
    __syncthreads();

    // streaming top-9: 2 threads per row, 64 columns each
    if (tid < 2 * MT) {
      const float* drow = sD + srow * NT + shalf * 64;
      const float* nbp  = snB + shalf * 64;
      for (int c = 0; c < 64; ++c) {
        float d2 = myNa + nbp[c] - 2.0f * drow[c];
        float d  = sqrtf(fmaxf(d2, 0.0f));
        if (d < best[KTOP - 1]) {
          best[KTOP - 1] = d;
#pragma unroll
          for (int j = KTOP - 1; j >= 1; --j) {
            float x = best[j - 1], y = best[j];
            best[j - 1] = fminf(x, y);
            best[j]     = fmaxf(x, y);
          }
        }
      }
    }
    __syncthreads();
  }

  // ---- final merge of the two sorted per-row partials ----
  if (tid >= MT && tid < 2 * MT) {
    float* p = sD + (tid - MT) * KTOP;
#pragma unroll
    for (int j = 0; j < KTOP; ++j) p[j] = best[j];
  }
  __syncthreads();
  if (tid < MT) {
    float other[KTOP];
#pragma unroll
    for (int j = 0; j < KTOP; ++j) other[j] = sD[tid * KTOP + j];
    // merged[k] = (k+1)-th smallest of union of two sorted 9-lists:
    //   min over i+j=k+1 of max(best[i-1], other[j-1])   (constant idx)
    float merged[KTOP];
#pragma unroll
    for (int k = 0; k < KTOP; ++k) {
      float m = 3.0e38f;
#pragma unroll
      for (int i = 0; i <= k + 1; ++i) {
        const int jj = k + 1 - i;
        const float av = (i == 0) ? -3.0e38f : best[i - 1];
        const float bv = (jj == 0) ? -3.0e38f : other[jj - 1];
        m = fminf(m, fmaxf(av, bv));
      }
      merged[k] = m;
    }
    float* o = topkOut + (size_t)(m0 + tid) * KTOP;
#pragma unroll
    for (int j = 0; j < KTOP; ++j) o[j] = merged[j];
  }
}

// ------------------------------------------------------------------
// K3: per-batch score: argmax over patches of NN distance, exp weight.
// ------------------------------------------------------------------
__global__ __launch_bounds__(256)
void finalize_score_kernel(const float* __restrict__ topk, float* __restrict__ out) {
  __shared__ float sv[256];
  __shared__ int   si[256];
  const int b = blockIdx.x, tid = threadIdx.x;
  float m = -3.0e38f; int mi = 0;
  for (int i = tid; i < HW; i += 256) {
    float v = topk[((size_t)b * HW + i) * KTOP];
    if (v > m) { m = v; mi = i; }
  }
  sv[tid] = m; si[tid] = mi;
  __syncthreads();
  for (int s = 128; s > 0; s >>= 1) {
    if (tid < s && sv[tid + s] > sv[tid]) { sv[tid] = sv[tid + s]; si[tid] = si[tid + s]; }
    __syncthreads();
  }
  if (tid == 0) {
    const size_t r = (size_t)b * HW + si[0];
    float mx = -3.0e38f, sum = 0.0f;
#pragma unroll
    for (int j = 0; j < KTOP; ++j) {
      float e = expf(topk[r * KTOP + j]);
      mx = fmaxf(mx, e);
      sum += e;
    }
    out[b] = sv[0] * (1.0f - mx / sum);
  }
}

// ------------------------------------------------------------------
// K4: bilinear 32x32 -> 256x256 (half-pixel centers, edge clamp).
// Source value = topk[row*9] (nearest-neighbor distance).
// ------------------------------------------------------------------
__global__ __launch_bounds__(256)
void resize_bilinear_kernel(const float* __restrict__ topk, float* __restrict__ out) {
  const int idx = blockIdx.x * 256 + threadIdx.x;
  const int x = idx & 255, y = (idx >> 8) & 255, b = idx >> 16;
  const float sy = (y + 0.5f) * 0.125f - 0.5f;
  const float sx = (x + 0.5f) * 0.125f - 0.5f;
  const float fy0 = floorf(sy), fx0 = floorf(sx);
  const int   y0 = (int)fy0,    x0 = (int)fx0;
  const float wy = sy - fy0,    wx = sx - fx0;
  const int y0c = imin(imax(y0, 0), 31),     y1c = imin(imax(y0 + 1, 0), 31);
  const int x0c = imin(imax(x0, 0), 31),     x1c = imin(imax(x0 + 1, 0), 31);
#define MASK_AT(iy, ix) topk[(((size_t)b * HW) + (size_t)(iy) * 32 + (ix)) * KTOP]
  const float v00 = MASK_AT(y0c, x0c), v01 = MASK_AT(y0c, x1c);
  const float v10 = MASK_AT(y1c, x0c), v11 = MASK_AT(y1c, x1c);
#undef MASK_AT
  const float v0 = v00 + (v01 - v00) * wx;
  const float v1 = v10 + (v11 - v10) * wx;
  out[idx] = v0 + (v1 - v0) * wy;
}

// ------------------------------------------------------------------
// K5/K6: separable 33-tap Gaussian (sigma=4), reflect-101 borders.
// ------------------------------------------------------------------
__global__ __launch_bounds__(256)
void blur_v_kernel(const float* __restrict__ in, float* __restrict__ out) {
  const int idx = blockIdx.x * 256 + threadIdx.x;
  const int x = idx & 255, y = (idx >> 8) & 255, b = idx >> 16;
  const float* img = in + (size_t)b * (IMG * IMG);
  float s = 0.0f, acc = 0.0f;
#pragma unroll
  for (int j = 0; j < 33; ++j) {
    const float d = (float)(j - 16);
    const float g = expf(-d * d * (1.0f / 32.0f));
    s += g;
    acc += g * img[refl101(y + j - 16, IMG) * IMG + x];
  }
  out[idx] = acc / s;
}

__global__ __launch_bounds__(256)
void blur_h_kernel(const float* __restrict__ in, float* __restrict__ out) {
  const int idx = blockIdx.x * 256 + threadIdx.x;
  const int x = idx & 255, y = (idx >> 8) & 255, b = idx >> 16;
  const float* row = in + (size_t)b * (IMG * IMG) + (size_t)y * IMG;
  float s = 0.0f, acc = 0.0f;
#pragma unroll
  for (int j = 0; j < 33; ++j) {
    const float d = (float)(j - 16);
    const float g = expf(-d * d * (1.0f / 32.0f));
    s += g;
    acc += g * row[refl101(x + j - 16, IMG)];
  }
  out[idx] = acc / s;
}

// ------------------------------------------------------------------
// Host orchestration
// ------------------------------------------------------------------
extern "C" void kernel_launch(void* const* d_in, const int* in_sizes, int n_in,
                              void* d_out, int out_size, void* d_ws, size_t ws_size,
                              hipStream_t stream) {
  (void)in_sizes; (void)n_in; (void)out_size; (void)ws_size;

  const float* inputs = (const float*)d_in[0];  // [8,32,32,512] -> [8192,512]
  const float* featv  = (const float*)d_in[1];  // [16384,512]
  float* out = (float*)d_out;                   // [8] score ++ [8*256*256] mask

  // workspace layout (bytes)
  char* ws = (char*)d_ws;
  __bf16* fA = (__bf16*)(ws);                                        //  8 MB
  __bf16* fB = (__bf16*)(ws + (size_t)MR * CD * 2);                  // 16 MB
  float*  nA = (float*)(ws + (size_t)MR * CD * 2 + (size_t)NR * CD * 2);
  float*  nB = nA + MR;
  float*  topk    = nB + NR;                                         // 8192*9
  float*  resized = topk + (size_t)MR * KTOP;                        //  2 MB
  float*  tmp     = resized + (size_t)BATCH * IMG * IMG;             //  2 MB

  cvt_norm_kernel<<<MR, 256, 0, stream>>>(inputs, fA, nA);
  cvt_norm_kernel<<<NR, 256, 0, stream>>>(featv,  fB, nB);
  fused_dist_topk_kernel<<<MR / MT, 256, 0, stream>>>(fA, fB, nA, nB, topk);
  finalize_score_kernel<<<BATCH, 256, 0, stream>>>(topk, out);
  resize_bilinear_kernel<<<(BATCH * IMG * IMG) / 256, 256, 0, stream>>>(topk, resized);
  blur_v_kernel<<<(BATCH * IMG * IMG) / 256, 256, 0, stream>>>(resized, tmp);
  blur_h_kernel<<<(BATCH * IMG * IMG) / 256, 256, 0, stream>>>(tmp, out + BATCH);
}
